// SelfAttention_48180943126565
// MI455X (gfx1250) — matrix operand order repaired and checked by reading, not compiled, not run on hardware
//
#include <hip/hip_runtime.h>

#define N_TOK 8192
#define C_DIM 1024
#define HD_DIM 128

typedef __bf16 bf16_t;
typedef __attribute__((ext_vector_type(16))) __bf16 v16bf;
typedef __attribute__((ext_vector_type(8)))  __bf16 v8bf;
typedef __attribute__((ext_vector_type(8)))  float  v8f;

// ---------------------------------------------------------------------------
// WMMA fragment helpers (layouts per CDNA5 ISA 7.12.2). wave32 only.
// ---------------------------------------------------------------------------

// A-matrix 16x32 (MxK), memory row-major: base[m*ld + k]
__device__ inline v16bf frag_a_rowmajor(const bf16_t* __restrict__ base, int ld) {
  int lane = threadIdx.x & 31;
  int m = lane & 15;
  int khalf = (lane >> 4) << 3;   // lanes 16-31 hold K+8 block
  v16bf f;
#pragma unroll
  for (int j = 0; j < 8; ++j) {
    int k = ((j & 3) << 1) + ((j >> 2) << 4) + khalf;
    const bf16_t* p = base + m * ld + k;
    f[2 * j]     = p[0];
    f[2 * j + 1] = p[1];
  }
  return f;
}

// B-matrix 32x16 (KxN), element B[k][n] stored at base[n*ld + k] (k contiguous).
__device__ inline v16bf frag_b_kfast(const bf16_t* __restrict__ base, int ld) {
  int lane = threadIdx.x & 31;
  int n = lane & 15;
  int koff = (lane >> 4) << 3;    // lanes 16-31 hold K+8 block
  v16bf f;
#pragma unroll
  for (int j = 0; j < 8; ++j) {
    int k = ((j & 3) << 1) + ((j >> 2) << 4) + koff;
    const bf16_t* p = base + (size_t)n * ld + k;
    f[2 * j]     = p[0];
    f[2 * j + 1] = p[1];
  }
  return f;
}

__device__ inline v8f wmma_bf16(v16bf a, v16bf b, v8f c) {
  return __builtin_amdgcn_wmma_f32_16x16x32_bf16(
      false, a, false, b, (short)0, c, false, false);
}

// ---------------------------------------------------------------------------
// f32 -> bf16 cast
// ---------------------------------------------------------------------------
__global__ void cast_f32_to_bf16(const float* __restrict__ in,
                                 bf16_t* __restrict__ out, int n) {
  int i = blockIdx.x * blockDim.x + threadIdx.x;
  if (i < n) out[i] = (bf16_t)in[i];
}

// ---------------------------------------------------------------------------
// GEMM: out(MxNout) = A(MxK, bf16) @ W(NoutxK, bf16)^T + bias(f32)
// block tile 64x64, 256 threads = 8 waves, each wave: 16 rows x 32 cols.
// mode: 0 = f32 row-major, 1 = bf16 row-major, 2 = bf16 TRANSPOSED (out[col*M+row])
// ---------------------------------------------------------------------------
__global__ void __launch_bounds__(256)
gemm_wmma_bias(const bf16_t* __restrict__ A, const bf16_t* __restrict__ W,
               const float* __restrict__ bias, void* __restrict__ out,
               int M, int Nout, int K, int mode) {
  int lane = threadIdx.x & 31;
  int wave = threadIdx.x >> 5;
  int wr = wave >> 1;
  int wc = wave & 1;
  int row0 = blockIdx.y * 64 + wr * 16;
  int col0 = blockIdx.x * 64 + wc * 32;

  v8f acc0 = {};
  v8f acc1 = {};
  for (int kk = 0; kk < K; kk += 32) {
    v16bf a  = frag_a_rowmajor(A + (size_t)row0 * K + kk, K);
    v16bf b0 = frag_b_kfast(W + (size_t)col0 * K + kk, K);
    v16bf b1 = frag_b_kfast(W + (size_t)(col0 + 16) * K + kk, K);
    acc0 = wmma_bf16(a, b0, acc0);
    acc1 = wmma_bf16(a, b1, acc1);
  }

  int n = lane & 15;
  int m0 = (lane >> 4) << 3;

  if (mode == 2) {
    // Transposed bf16 store: lane n owns a fixed output column; its 8 row
    // values are contiguous in the transposed layout -> one 16B store each.
    v8bf t0, t1;
#pragma unroll
    for (int r = 0; r < 8; ++r) {
      t0[r] = (bf16_t)(acc0[r] + bias[col0 + n]);
      t1[r] = (bf16_t)(acc1[r] + bias[col0 + 16 + n]);
    }
    bf16_t* o = (bf16_t*)out;
    *(v8bf*)(o + (size_t)(col0 + n) * M + row0 + m0)      = t0;
    *(v8bf*)(o + (size_t)(col0 + 16 + n) * M + row0 + m0) = t1;
    return;
  }

#pragma unroll
  for (int r = 0; r < 8; ++r) {
    int row = row0 + m0 + r;
    float c0 = acc0[r] + bias[col0 + n];
    float c1 = acc1[r] + bias[col0 + 16 + n];
    if (mode == 1) {
      bf16_t* o = (bf16_t*)out;
      o[(size_t)row * Nout + col0 + n]      = (bf16_t)c0;
      o[(size_t)row * Nout + col0 + 16 + n] = (bf16_t)c1;
    } else {
      float* o = (float*)out;
      o[(size_t)row * Nout + col0 + n]      = c0;
      o[(size_t)row * Nout + col0 + 16 + n] = c1;
    }
  }
}

// ---------------------------------------------------------------------------
// Flash attention: one block = 32 query rows, 8 waves.
// Per 128-key tile: wave w computes scores for keys [16w,16w+16) against both
// 16-query halves (8 WMMAs), block-wide online softmax in f32, p(bf16)->LDS,
// then each wave does p @ V for channel slice [128w,128w+128): each V
// B-fragment is reused by TWO WMMAs (one per query half) -> 2x arithmetic
// intensity on the dominant GEMM. V is TRANSPOSED (VT: C_DIM x N_TOK).
// Q tile is fetched with global_load_async_to_lds_b128 (ASYNCcnt path).
// ---------------------------------------------------------------------------
__global__ void __launch_bounds__(256)
flash_attn_wmma(const bf16_t* __restrict__ Q, const bf16_t* __restrict__ Kg,
                const bf16_t* __restrict__ VT, bf16_t* __restrict__ O) {
  __shared__ bf16_t qs[32 * HD_DIM];     // query tile 32 x 128 (8 KB)
  __shared__ bf16_t pbuf[32 * 128];      // probabilities 32 x 128-key tile (8 KB)
  __shared__ float wmax[8][32];
  __shared__ float wsum[8][32];
  __shared__ float mrun[32], lrun[32], mnew[32], cf[32];

  const int tid  = threadIdx.x;
  const int lane = tid & 31;
  const int wave = tid >> 5;
  const int q0   = blockIdx.x * 32;
  const int n    = lane & 15;
  const int m0   = (lane >> 4) << 3;
  const float scale = 0.08838834764831845f;  // 1/sqrt(128)

  // --- async copy of the contiguous 32x128 Q tile into LDS (2x b128/thread) ---
  {
    const bf16_t* g0 = Q + (size_t)q0 * HD_DIM + tid * 8;
    const bf16_t* g1 = g0 + 256 * 8;
    unsigned l0 = (unsigned)(size_t)(qs + tid * 8);
    unsigned l1 = l0 + 256 * 8 * 2;
    asm volatile(
        "global_load_async_to_lds_b128 %0, %1, off\n\t"
        "global_load_async_to_lds_b128 %2, %3, off"
        :: "v"(l0), "v"(g0), "v"(l1), "v"(g1) : "memory");
    asm volatile("s_wait_asynccnt 0" ::: "memory");
  }
  if (tid < 32) { mrun[tid] = -1e30f; lrun[tid] = 0.0f; }
  __syncthreads();

  v8f acc[2][8];
#pragma unroll
  for (int h = 0; h < 2; ++h)
#pragma unroll
    for (int i = 0; i < 8; ++i) acc[h][i] = (v8f){};

  for (int kt = 0; kt < N_TOK; kt += 128) {
    const int key0 = kt + wave * 16;

    // Prefetch next key tile of K (global_prefetch_b8).
    if (kt + 128 < N_TOK) {
      __builtin_prefetch(Kg + (size_t)(key0 + 128) * HD_DIM + lane * 64, 0, 1);
    }

    // --- scores s(2 x 16q x 16k) = q @ k^T over HD=128 ---
    v8f s[2] = {(v8f){}, (v8f){}};
#pragma unroll
    for (int ks = 0; ks < 4; ++ks) {
      v16bf b = frag_b_kfast(Kg + (size_t)key0 * HD_DIM + ks * 32, HD_DIM);
      v16bf a0 = frag_a_rowmajor(qs + ks * 32, HD_DIM);
      v16bf a1 = frag_a_rowmajor(qs + 16 * HD_DIM + ks * 32, HD_DIM);
      s[0] = wmma_bf16(a0, b, s[0]);
      s[1] = wmma_bf16(a1, b, s[1]);
    }

    // --- per-row max of this wave's 16-key tile ---
    float sv[2][8];
#pragma unroll
    for (int h = 0; h < 2; ++h) {
#pragma unroll
      for (int r = 0; r < 8; ++r) {
        float v = s[h][r] * scale;
        sv[h][r] = v;
        float m = v;
        m = fmaxf(m, __shfl_xor(m, 1));
        m = fmaxf(m, __shfl_xor(m, 2));
        m = fmaxf(m, __shfl_xor(m, 4));
        m = fmaxf(m, __shfl_xor(m, 8));
        if (n == 0) wmax[wave][h * 16 + m0 + r] = m;
      }
    }
    __syncthreads();  // B1: wmax visible

    if (tid < 32) {
      float m_old = mrun[tid];
      float tm = wmax[0][tid];
#pragma unroll
      for (int w = 1; w < 8; ++w) tm = fmaxf(tm, wmax[w][tid]);
      float mn = fmaxf(m_old, tm);
      mnew[tid] = mn;
      cf[tid]   = __expf(m_old - mn);
      mrun[tid] = mn;
    }
    __syncthreads();  // B2: mnew, cf visible

    // --- p = exp(s - mnew), partial row sums, acc rescale, p -> LDS ---
#pragma unroll
    for (int h = 0; h < 2; ++h) {
#pragma unroll
      for (int r = 0; r < 8; ++r) {
        int row = h * 16 + m0 + r;
        float p = __expf(sv[h][r] - mnew[row]);
        float su = p;
        su += __shfl_xor(su, 1);
        su += __shfl_xor(su, 2);
        su += __shfl_xor(su, 4);
        su += __shfl_xor(su, 8);
        if (n == 0) wsum[wave][row] = su;
        pbuf[row * 128 + wave * 16 + n] = (bf16_t)p;
        float c = cf[row];
#pragma unroll
        for (int i = 0; i < 8; ++i) acc[h][i][r] *= c;
      }
    }
    __syncthreads();  // B3: pbuf, wsum visible

    if (tid < 32) {
      float l = lrun[tid] * cf[tid];
#pragma unroll
      for (int w = 0; w < 8; ++w) l += wsum[w][tid];
      lrun[tid] = l;
    }

    // --- acc += p(32x128) @ V(128 x 128-col slice); each B frag used twice ---
#pragma unroll
    for (int ks = 0; ks < 4; ++ks) {
      v16bf a0 = frag_a_rowmajor(pbuf + ks * 32, 128);
      v16bf a1 = frag_a_rowmajor(pbuf + 16 * 128 + ks * 32, 128);
#pragma unroll
      for (int i = 0; i < 8; ++i) {
        v16bf b = frag_b_kfast(
            VT + (size_t)(wave * 128 + i * 16) * N_TOK + kt + ks * 32, N_TOK);
        acc[0][i] = wmma_bf16(a0, b, acc[0][i]);
        acc[1][i] = wmma_bf16(a1, b, acc[1][i]);
      }
    }
  }

  __syncthreads();
#pragma unroll
  for (int h = 0; h < 2; ++h) {
#pragma unroll
    for (int r = 0; r < 8; ++r) {
      int row = h * 16 + m0 + r;
      float inv = 1.0f / lrun[row];
#pragma unroll
      for (int i = 0; i < 8; ++i) {
        O[(size_t)(q0 + row) * C_DIM + wave * 128 + i * 16 + n] =
            (bf16_t)(acc[h][i][r] * inv);
      }
    }
  }
}

// ---------------------------------------------------------------------------
// Host launcher
// ---------------------------------------------------------------------------
extern "C" void kernel_launch(void* const* d_in, const int* in_sizes, int n_in,
                              void* d_out, int out_size, void* d_ws, size_t ws_size,
                              hipStream_t stream) {
  const float* x  = (const float*)d_in[0];
  const float* Wq = (const float*)d_in[1];
  const float* bq = (const float*)d_in[2];
  const float* Wk = (const float*)d_in[3];
  const float* bk = (const float*)d_in[4];
  const float* Wv = (const float*)d_in[5];
  const float* bv = (const float*)d_in[6];
  const float* Wo = (const float*)d_in[7];
  const float* bo = (const float*)d_in[8];
  float* out = (float*)d_out;

  char* ws = (char*)d_ws;
  size_t off = 0;
  auto alloc = [&](size_t bytes) {
    char* p = ws + off;
    off += (bytes + 255) & ~(size_t)255;
    return p;
  };
  bf16_t* xb   = (bf16_t*)alloc((size_t)N_TOK * C_DIM * 2);   // x in bf16
  bf16_t* wqb  = (bf16_t*)alloc((size_t)HD_DIM * C_DIM * 2);
  bf16_t* wkb  = (bf16_t*)alloc((size_t)HD_DIM * C_DIM * 2);
  bf16_t* wvb  = (bf16_t*)alloc((size_t)C_DIM * C_DIM * 2);
  bf16_t* wob  = (bf16_t*)alloc((size_t)C_DIM * C_DIM * 2);
  bf16_t* qb   = (bf16_t*)alloc((size_t)N_TOK * HD_DIM * 2);
  bf16_t* kb   = (bf16_t*)alloc((size_t)N_TOK * HD_DIM * 2);
  bf16_t* vtb  = (bf16_t*)alloc((size_t)N_TOK * C_DIM * 2);   // V transposed (C x N)
  bf16_t* ob   = xb;  // attention output reuses xb (x dead after QKV gemms)

  const int T = 256;
  cast_f32_to_bf16<<<(N_TOK * C_DIM + T - 1) / T, T, 0, stream>>>(x, xb, N_TOK * C_DIM);
  cast_f32_to_bf16<<<(HD_DIM * C_DIM + T - 1) / T, T, 0, stream>>>(Wq, wqb, HD_DIM * C_DIM);
  cast_f32_to_bf16<<<(HD_DIM * C_DIM + T - 1) / T, T, 0, stream>>>(Wk, wkb, HD_DIM * C_DIM);
  cast_f32_to_bf16<<<(C_DIM * C_DIM + T - 1) / T, T, 0, stream>>>(Wv, wvb, C_DIM * C_DIM);
  cast_f32_to_bf16<<<(C_DIM * C_DIM + T - 1) / T, T, 0, stream>>>(Wo, wob, C_DIM * C_DIM);

  // q = x @ Wq^T + bq  (bf16 row-major), k likewise
  gemm_wmma_bias<<<dim3(HD_DIM / 64, N_TOK / 64), 256, 0, stream>>>(
      xb, wqb, bq, qb, N_TOK, HD_DIM, C_DIM, 1);
  gemm_wmma_bias<<<dim3(HD_DIM / 64, N_TOK / 64), 256, 0, stream>>>(
      xb, wkb, bk, kb, N_TOK, HD_DIM, C_DIM, 1);
  // v = x @ Wv^T + bv, stored TRANSPOSED (C x N) for contiguous B-fragments
  gemm_wmma_bias<<<dim3(C_DIM / 64, N_TOK / 64), 256, 0, stream>>>(
      xb, wvb, bv, vtb, N_TOK, C_DIM, C_DIM, 2);

  // attn(q,k,v) -> ob (bf16, row-major N x C)
  flash_attn_wmma<<<N_TOK / 32, 256, 0, stream>>>(qb, kb, vtb, ob);

  // out = ob @ Wo^T + bo (f32)
  gemm_wmma_bias<<<dim3(C_DIM / 64, N_TOK / 64), 256, 0, stream>>>(
      ob, wob, bo, out, N_TOK, C_DIM, C_DIM, 0);
}
